// MultiHeadAttention_13675175870663
// MI455X (gfx1250) — matrix operand order repaired and checked by reading, not compiled, not run on hardware
//
#include <hip/hip_runtime.h>
#include <hip/hip_bf16.h>

#define D_MODEL  1024
#define N_HEADS  16
#define HEAD_DIM 64
#define BATCH    2
#define SEQ      2048

typedef __attribute__((ext_vector_type(16))) __bf16 v16bf;
typedef __attribute__((ext_vector_type(8)))  __bf16 v8bf;
typedef __attribute__((ext_vector_type(8)))  float  v8f;
typedef __attribute__((ext_vector_type(4)))  float  v4f;

typedef unsigned int u32x4 __attribute__((ext_vector_type(4)));
typedef int          i32x4 __attribute__((ext_vector_type(4)));
typedef int          i32x8 __attribute__((ext_vector_type(8)));

__device__ __forceinline__ __bf16 f2bf(float f) {
  unsigned u = __builtin_bit_cast(unsigned, f);
  unsigned r = (u + 0x7FFFu + ((u >> 16) & 1u)) >> 16;
  return __builtin_bit_cast(__bf16, (unsigned short)r);
}

__device__ __forceinline__ v16bf cat8(v8bf lo, v8bf hi) {
  return __builtin_shufflevector(lo, hi, 0, 1, 2, 3, 4, 5, 6, 7,
                                         8, 9, 10, 11, 12, 13, 14, 15);
}

// ---------------------------------------------------------------------------
// TDM 2D tile load: global (row-major, 2-byte elems) -> LDS, async.
// D# bitfields per CDNA5 ISA ch.8. Tracked by TENSORcnt.
// ---------------------------------------------------------------------------
__device__ __forceinline__ void tdm_load_2d(unsigned lds_off, const void* gaddr,
                                            unsigned tile_d0, unsigned tile_d1,
                                            unsigned long long stride0) {
  const unsigned long long ga = (unsigned long long)(size_t)gaddr;
  u32x4 g0;
  g0.x = 1u;                                                // count=1, user mode
  g0.y = lds_off;                                           // lds_addr
  g0.z = (unsigned)ga;                                      // global_addr[31:0]
  g0.w = (unsigned)((ga >> 32) & 0x1FFFFFFu) | (2u << 30);  // addr[56:32] | type=2
  i32x8 g1;
  g1[0] = (int)(1u << 16);                                  // data_size=2B, mask=0
  g1[1] = (int)((tile_d0 & 0xFFFFu) << 16);                 // tensor_dim0[15:0]
  g1[2] = (int)((tile_d0 >> 16) | ((tile_d1 & 0xFFFFu) << 16));  // dim0 hi|dim1 lo
  g1[3] = (int)((tile_d1 >> 16) | (tile_d0 << 16));         // dim1 hi | tile_dim0
  g1[4] = (int)(tile_d1 & 0xFFFFu);                         // tile_dim1, tile_dim2=0
  g1[5] = (int)(unsigned)(stride0 & 0xFFFFFFFFull);         // dim0_stride[31:0]
  g1[6] = (int)(unsigned)(stride0 >> 32);                   // dim0_stride[47:32]
  g1[7] = 0;
  i32x4 gz; gz[0] = 0; gz[1] = 0; gz[2] = 0; gz[3] = 0;
#if __clang_major__ >= 23
  i32x8 gz8 = g1 ^ g1;
  __builtin_amdgcn_tensor_load_to_lds(g0, g1, gz, gz, gz8, 0);
#else
  __builtin_amdgcn_tensor_load_to_lds(g0, g1, gz, gz, 0);
#endif
}

__device__ __forceinline__ unsigned lds_off_of(const void* p) {
  return (unsigned)(size_t)p;  // low 32 bits of flat LDS address = LDS offset
}

// ---------------------------------------------------------------------------
// fp32 -> bf16 (RNE) elementwise conversion
// ---------------------------------------------------------------------------
__global__ void cvt_f32_bf16(const float* __restrict__ src,
                             __bf16* __restrict__ dst, int n) {
  int i = blockIdx.x * blockDim.x + threadIdx.x;
  if (i < n) dst[i] = f2bf(src[i]);
}

// ---------------------------------------------------------------------------
// Y = X(MxK,bf16) * W(NxK,bf16)^T + bias(fp32)
// MODE 0: bf16 out, (B, H, S, Dh)     (Q, K)
// MODE 1: fp32 out, flat (M x N)      (final O-proj)
// MODE 2: bf16 out, (B, H, Dh, S)     (V, transposed for contiguous P*V)
// Block: 256 thr = 8 waves; wave tile 16x64; block tile 128x64.
// X/W tiles DMA'd into double-buffered LDS by the TDM; waves consume via WMMA.
// ---------------------------------------------------------------------------
template <int MODE>
__global__ void gemm_wmma(const __bf16* __restrict__ X,
                          const __bf16* __restrict__ W,
                          const float* __restrict__ bias,
                          void* __restrict__ out,
                          int M, int N, int K) {
  __shared__ __align__(32) __bf16 s_x[2][128][32];  // 16 KB ping-pong
  __shared__ __align__(32) __bf16 s_w[2][64][32];   // 8 KB ping-pong

  const int tid   = threadIdx.x;
  const int lane  = tid & 31;
  const int wave  = tid >> 5;
  const int l16   = lane & 15;
  const int lhalf = lane >> 4;

  const int m0 = blockIdx.x * 128;
  const int n0 = blockIdx.y * 64;

  v8f acc[4] = {};
  const int niter = K / 32;

  // Prologue: TDM prefetch of first k-chunk into buffer 0.
  if (wave == 0) {
    tdm_load_2d(lds_off_of(&s_x[0][0][0]), X + (size_t)m0 * K, 32u, 128u,
                (unsigned long long)K);
    tdm_load_2d(lds_off_of(&s_w[0][0][0]), W + (size_t)n0 * K, 32u, 64u,
                (unsigned long long)K);
  }

  for (int it = 0; it < niter; ++it) {
    const int cur = it & 1;
    if (wave == 0) __builtin_amdgcn_s_wait_tensorcnt(0);
    __syncthreads();  // tiles in buf[cur] are visible to all waves

    // Kick off DMA of the next k-chunk into the other buffer.
    if (wave == 0 && it + 1 < niter) {
      const int k1 = (it + 1) * 32;
      tdm_load_2d(lds_off_of(&s_x[cur ^ 1][0][0]), X + (size_t)m0 * K + k1,
                  32u, 128u, (unsigned long long)K);
      tdm_load_2d(lds_off_of(&s_w[cur ^ 1][0][0]), W + (size_t)n0 * K + k1,
                  32u, 64u, (unsigned long long)K);
    }

    // ---- A fragment from LDS: rows wave*16 + l16 ----
    v16bf a;
    {
      const __bf16* ap = &s_x[cur][wave * 16 + l16][lhalf * 16];
      a = cat8(*(const v8bf*)ap, *(const v8bf*)(ap + 8));
    }
#pragma unroll
    for (int s = 0; s < 4; ++s) {
      v16bf b = *(const v16bf*)&s_w[cur][s * 16 + l16][lhalf * 16];
      acc[s] = __builtin_amdgcn_wmma_f32_16x16x32_bf16(
          false, a, false, b, (short)0, acc[s], false, false);
    }
    __syncthreads();  // done reading buf[cur] before it is DMA'd again
  }

  const int mrow0 = m0 + wave * 16;
#pragma unroll
  for (int s = 0; s < 4; ++s) {
#pragma unroll
    for (int i = 0; i < 8; ++i) {
      const int m = mrow0 + i + lhalf * 8;
      const int n = n0 + s * 16 + l16;
      float v = acc[s][i] + bias[n];
      if (MODE == 0) {
        const int bb = m / SEQ, ss = m % SEQ;
        const int h = n / HEAD_DIM, dh = n % HEAD_DIM;
        ((__bf16*)out)[(((size_t)(bb * N_HEADS + h) * SEQ) + ss) * HEAD_DIM + dh] =
            f2bf(v);
      } else if (MODE == 2) {
        const int bb = m / SEQ, ss = m % SEQ;
        const int h = n / HEAD_DIM, dh = n % HEAD_DIM;
        ((__bf16*)out)[(((size_t)(bb * N_HEADS + h) * HEAD_DIM) + dh) * SEQ + ss] =
            f2bf(v);
      } else {
        ((float*)out)[(size_t)m * N + n] = v;
      }
    }
  }
}

// ---------------------------------------------------------------------------
// Flash attention. Q,K in (B,H,S,Dh); V pre-transposed to (B,H,Dh,S).
// Block: 256 thr = 8 waves = 128 query rows of one (b,h).
// K tile (32x64) and V^T tile (64x32) DMA'd by the TDM into ping-pong LDS.
// Per-wave online softmax staged through LDS; all GEMM work in bf16 WMMA.
// ---------------------------------------------------------------------------
__global__ void flash_attn(const __bf16* __restrict__ Q,
                           const __bf16* __restrict__ Kt,
                           const __bf16* __restrict__ Vt,
                           __bf16* __restrict__ attn) {
  __shared__ __align__(32) __bf16 s_k[2][32][64];   // [key][dim]  8 KB
  __shared__ __align__(32) __bf16 s_v[2][64][32];   // [dim][key]  8 KB
  __shared__ __align__(16) float  s_sc[8][16][32];  // 16 KB
  __shared__ __align__(16) __bf16 s_p[8][16][32];   // 8 KB
  __shared__ float s_m[8][16];
  __shared__ float s_l[8][16];
  __shared__ float s_f[8][16];

  const int tid   = threadIdx.x;
  const int lane  = tid & 31;
  const int wave  = tid >> 5;
  const int l16   = lane & 15;
  const int lhalf = lane >> 4;

  const int blk = blockIdx.x;
  const int qt  = blk & 15;          // SEQ/128 query tiles
  const int h   = (blk >> 4) & 15;   // head
  const int bb  = blk >> 8;          // batch
  const int q0  = qt * 128 + wave * 16;

  const __bf16* Qh = Q  + (size_t)(bb * N_HEADS + h) * SEQ * HEAD_DIM;
  const __bf16* Kh = Kt + (size_t)(bb * N_HEADS + h) * SEQ * HEAD_DIM;
  const __bf16* Vh = Vt + (size_t)(bb * N_HEADS + h) * HEAD_DIM * SEQ;

  // Q fragments (two k=32 chunks covering Dh=64), loaded once
  v16bf qa[2];
  {
    const __bf16* qrow = Qh + (size_t)(q0 + l16) * HEAD_DIM + lhalf * 16;
#pragma unroll
    for (int c = 0; c < 2; ++c) {
      qa[c] = cat8(*(const v8bf*)(qrow + c * 32),
                   *(const v8bf*)(qrow + c * 32 + 8));
    }
  }

  v8f acc[4] = {};
  if (lane < 16) {
    s_m[wave][lane] = -3.0e38f;
    s_l[wave][lane] = 0.0f;
  }

  const float scale = 0.125f;          // 1/sqrt(64)
  const int ktend_blk = qt * 4 + 4;    // causal bound for the whole block

  // Prologue: TDM prefetch of key-tile 0 into buffer 0.
  if (wave == 0) {
    tdm_load_2d(lds_off_of(&s_k[0][0][0]), Kh, 64u, 32u, 64ull);
    tdm_load_2d(lds_off_of(&s_v[0][0][0]), Vh, 32u, 64u, (unsigned long long)SEQ);
  }

  for (int kt = 0; kt < ktend_blk; ++kt) {
    const int kk0 = kt * 32;
    const int cur = kt & 1;

    if (wave == 0) __builtin_amdgcn_s_wait_tensorcnt(0);
    __syncthreads();

    if (wave == 0 && kt + 1 < ktend_blk) {
      const int kn = (kt + 1) * 32;
      tdm_load_2d(lds_off_of(&s_k[cur ^ 1][0][0]),
                  Kh + (size_t)kn * HEAD_DIM, 64u, 32u, 64ull);
      tdm_load_2d(lds_off_of(&s_v[cur ^ 1][0][0]),
                  Vh + kn, 32u, 64u, (unsigned long long)SEQ);
    }

    if (kk0 <= q0 + 15) {   // wave-uniform causal skip
      // ---- scores = Q * K^T (two 16-key subtiles) ----
#pragma unroll
      for (int s2 = 0; s2 < 2; ++s2) {
        const int keyl = s2 * 16 + l16;
        const int key  = kk0 + keyl;
        v8f c = {};
#pragma unroll
        for (int kc = 0; kc < 2; ++kc) {
          v16bf b = *(const v16bf*)&s_k[cur][keyl][kc * 32 + lhalf * 16];
          c = __builtin_amdgcn_wmma_f32_16x16x32_bf16(
              false, qa[kc], false, b, (short)0, c, false, false);
        }
#pragma unroll
        for (int i = 0; i < 8; ++i) {
          const int r = i + lhalf * 8;
          float v = c[i] * scale;
          if (key > q0 + r) v = -3.0e38f;  // causal mask
          s_sc[wave][r][keyl] = v;
        }
      }
      asm volatile("s_wait_dscnt 0" ::: "memory");

      // ---- online softmax: lanes 0-15 each own a row ----
      if (lane < 16) {
        const int r = lane;
        float mo = s_m[wave][r];
        v4f mxv = {mo, mo, mo, mo};
#pragma unroll
        for (int j = 0; j < 32; j += 4) {
          mxv = __builtin_elementwise_max(mxv, *(const v4f*)&s_sc[wave][r][j]);
        }
        float mx = fmaxf(fmaxf(mxv[0], mxv[1]), fmaxf(mxv[2], mxv[3]));
        const float fac = __expf(mo - mx);
        float sum = 0.0f;
#pragma unroll 8
        for (int j = 0; j < 32; ++j) {
          float p = __expf(s_sc[wave][r][j] - mx);
          sum += p;
          s_p[wave][r][j] = f2bf(p);
        }
        s_m[wave][r] = mx;
        s_l[wave][r] = s_l[wave][r] * fac + sum;
        s_f[wave][r] = fac;
      }
      asm volatile("s_wait_dscnt 0" ::: "memory");

      // ---- rescale running accumulators ----
#pragma unroll
      for (int s = 0; s < 4; ++s) {
#pragma unroll
        for (int i = 0; i < 8; ++i) {
          acc[s][i] *= s_f[wave][i + lhalf * 8];
        }
      }

      // ---- P fragment from LDS ----
      v16bf pfrag;
      {
        const __bf16* prow = &s_p[wave][l16][lhalf * 16];
        pfrag = cat8(*(const v8bf*)prow, *(const v8bf*)(prow + 8));
      }

      // ---- out += P * V (four 16-dim subtiles, contiguous from s_v) ----
#pragma unroll
      for (int s = 0; s < 4; ++s) {
        v16bf vb = *(const v16bf*)&s_v[cur][s * 16 + l16][lhalf * 16];
        acc[s] = __builtin_amdgcn_wmma_f32_16x16x32_bf16(
            false, pfrag, false, vb, (short)0, acc[s], false, false);
      }
    }
    __syncthreads();  // all reads of buf[cur] done before it is DMA'd again
  }

  // ---- finalize: divide by row sums, write (B, S, D) bf16 ----
#pragma unroll
  for (int s = 0; s < 4; ++s) {
#pragma unroll
    for (int i = 0; i < 8; ++i) {
      const int r = i + lhalf * 8;
      const float ov = acc[s][i] / s_l[wave][r];
      const int srow = q0 + r;
      const int dcol = h * HEAD_DIM + s * 16 + l16;
      attn[((size_t)bb * SEQ + srow) * D_MODEL + dcol] = f2bf(ov);
    }
  }
}

// ---------------------------------------------------------------------------
// Host-side orchestration
// ---------------------------------------------------------------------------
extern "C" void kernel_launch(void* const* d_in, const int* in_sizes, int n_in,
                              void* d_out, int out_size, void* d_ws,
                              size_t ws_size, hipStream_t stream) {
  (void)in_sizes; (void)n_in; (void)out_size; (void)ws_size;

  const float* x  = (const float*)d_in[0];
  // d_in[1] = causal mask: handled analytically in flash_attn
  const float* Wq = (const float*)d_in[2];
  const float* bq = (const float*)d_in[3];
  const float* Wk = (const float*)d_in[4];
  const float* bk = (const float*)d_in[5];
  const float* Wv = (const float*)d_in[6];
  const float* bv = (const float*)d_in[7];
  const float* Wo = (const float*)d_in[8];
  const float* bo = (const float*)d_in[9];
  float* out = (float*)d_out;

  const size_t nX = (size_t)BATCH * SEQ * D_MODEL;  // 4,194,304
  const size_t nW = (size_t)D_MODEL * D_MODEL;      // 1,048,576

  char* ws = (char*)d_ws;
  __bf16* xb    = (__bf16*)ws; ws += nX * sizeof(__bf16);
  __bf16* wqb   = (__bf16*)ws; ws += nW * sizeof(__bf16);
  __bf16* wkb   = (__bf16*)ws; ws += nW * sizeof(__bf16);
  __bf16* wvb   = (__bf16*)ws; ws += nW * sizeof(__bf16);
  __bf16* wob   = (__bf16*)ws; ws += nW * sizeof(__bf16);
  __bf16* Qb    = (__bf16*)ws; ws += nX * sizeof(__bf16);
  __bf16* Kb    = (__bf16*)ws; ws += nX * sizeof(__bf16);
  __bf16* Vb    = (__bf16*)ws; ws += nX * sizeof(__bf16);
  __bf16* attnb = (__bf16*)ws; ws += nX * sizeof(__bf16);

  // 1) fp32 -> bf16 conversions
  cvt_f32_bf16<<<dim3((unsigned)((nX + 255) / 256)), dim3(256), 0, stream>>>(
      x, xb, (int)nX);
  cvt_f32_bf16<<<dim3((unsigned)((nW + 255) / 256)), dim3(256), 0, stream>>>(
      Wq, wqb, (int)nW);
  cvt_f32_bf16<<<dim3((unsigned)((nW + 255) / 256)), dim3(256), 0, stream>>>(
      Wk, wkb, (int)nW);
  cvt_f32_bf16<<<dim3((unsigned)((nW + 255) / 256)), dim3(256), 0, stream>>>(
      Wv, wvb, (int)nW);
  cvt_f32_bf16<<<dim3((unsigned)((nW + 255) / 256)), dim3(256), 0, stream>>>(
      Wo, wob, (int)nW);

  // 2) Q/K/V projections (WMMA GEMMs); V written transposed (B,H,Dh,S)
  const dim3 ggrid(BATCH * SEQ / 128, D_MODEL / 64);  // (32, 16)
  gemm_wmma<0><<<ggrid, dim3(256), 0, stream>>>(xb, wqb, bq, Qb,
                                                BATCH * SEQ, D_MODEL, D_MODEL);
  gemm_wmma<0><<<ggrid, dim3(256), 0, stream>>>(xb, wkb, bk, Kb,
                                                BATCH * SEQ, D_MODEL, D_MODEL);
  gemm_wmma<2><<<ggrid, dim3(256), 0, stream>>>(xb, wvb, bv, Vb,
                                                BATCH * SEQ, D_MODEL, D_MODEL);

  // 3) fused causal flash attention
  flash_attn<<<dim3(BATCH * N_HEADS * (SEQ / 128)), dim3(256), 0, stream>>>(
      Qb, Kb, Vb, attnb);

  // 4) output projection -> fp32 d_out
  gemm_wmma<1><<<ggrid, dim3(256), 0, stream>>>(attnb, wob, bo, out,
                                                BATCH * SEQ, D_MODEL, D_MODEL);
}
